// InfoNCELoss_54700703482318
// MI455X (gfx1250) — matrix operand order repaired and checked by reading, not compiled, not run on hardware
//
#include <hip/hip_runtime.h>

// ---------------------------------------------------------------------------
// InfoNCE loss, fused E·E^T GEMM + streaming logsumexp on CDNA5 (gfx1250).
// B=4, N=4096, D=256, T=0.1.  bf16 WMMA (f32 accumulate), wave32.
// Column sweep split 4-ways per row tile; whole-tile B prefetch pinned with a
// sched_barrier; softmax update software-pipelined into the WMMA shadow.
// ---------------------------------------------------------------------------

typedef __attribute__((ext_vector_type(16))) __bf16 v16bf;
typedef __attribute__((ext_vector_type(8)))  __bf16 v8bf;
typedef __attribute__((ext_vector_type(8)))  float  v8f;

constexpr int   Bn     = 4;
constexpr int   Nn     = 4096;
constexpr int   Dn     = 256;
constexpr int   CSPLIT = 4;                 // column-sweep splits (waves/tile)
constexpr int   NCOLS  = Nn / CSPLIT;
constexpr float SCALE2 = 14.426950408889634f;   // (1/T) * log2(e)
constexpr float LN2    = 0.6931471805599453f;

union V16 { v16bf v; v8bf h[2]; };

__device__ __forceinline__ void sched_fence() {
#if __has_builtin(__builtin_amdgcn_sched_barrier)
  __builtin_amdgcn_sched_barrier(0);
#endif
}

__device__ inline float fast_exp2(float x) {    // v_exp_f32 is natively base-2
#if __has_builtin(__builtin_amdgcn_exp2f)
  return __builtin_amdgcn_exp2f(x);
#else
  return exp2f(x);
#endif
}

__device__ inline float fast_log2(float x) {    // v_log_f32 is natively base-2
#if __has_builtin(__builtin_amdgcn_logf)
  return __builtin_amdgcn_logf(x);
#else
  return log2f(x);
#endif
}

__device__ inline unsigned short f32_to_bf16(float f) {
  unsigned u = __builtin_bit_cast(unsigned, f);
  unsigned r = u + 0x7fffu + ((u >> 16) & 1u);   // round-to-nearest-even
  return (unsigned short)(r >> 16);
}

__device__ inline unsigned hash3(unsigned b, unsigned i, unsigned k) {
  unsigned h = b * 0x8da6b343u + i * 0xd8163841u + k * 0xcb1ab31fu;
  h ^= h >> 13; h *= 0x85ebca6bu; h ^= h >> 16;
  return h;
}

template <int PAT>
__device__ inline float swz(float x) {   // group-of-32 XOR swizzle (ds_swizzle_b32)
  return __builtin_bit_cast(float,
      __builtin_amdgcn_ds_swizzle(__builtin_bit_cast(int, x), PAT));
}

// -------------------------- 1) fp32 -> bf16 --------------------------------
__global__ __launch_bounds__(256)
void infonce_cvt_kernel(const float* __restrict__ e, unsigned short* __restrict__ o) {
  int idx = (blockIdx.x * 256 + threadIdx.x) * 4;
  float4 f = *(const float4*)(e + idx);
  ushort4 r;
  r.x = f32_to_bf16(f.x); r.y = f32_to_bf16(f.y);
  r.z = f32_to_bf16(f.z); r.w = f32_to_bf16(f.w);
  *(ushort4*)(o + idx) = r;
}

// ----------------- 2) deterministic positive per anchor --------------------
__global__ __launch_bounds__(256)
void infonce_pos_kernel(const int* __restrict__ groups, int* __restrict__ jpos) {
  __shared__ int gs[Nn];
  const int bpb = Nn / 256;                     // blocks per batch
  const int b = blockIdx.x / bpb;
  const int i = (blockIdx.x % bpb) * 256 + threadIdx.x;
  const int* g = groups + b * Nn;
  for (int k = threadIdx.x; k < Nn; k += 256) gs[k] = g[k];
  __syncthreads();
  const int gi = gs[i];
  unsigned best = 0; int bj = -1;
  for (int k = 0; k < Nn; ++k) {
    unsigned h = hash3(b, i, k);
    bool ok = (gs[k] == gi) & (k != i);
    if (ok && (bj < 0 || h > best)) { best = h; bj = k; }
  }
  jpos[b * Nn + i] = bj;
}

// --------- 3) fused bf16-WMMA GEMM + online logsumexp (exp2 domain) --------
__global__ __launch_bounds__(128)
void infonce_gemm_kernel(const unsigned short* __restrict__ Ebf,
                         const int* __restrict__ jpos,
                         float* __restrict__ lossw) {
  __shared__ float sm[CSPLIT][16];
  __shared__ float ss[CSPLIT][16];
  __shared__ float sp[CSPLIT][16];

  const int lane = threadIdx.x & 31;
  const int wave = threadIdx.x >> 5;            // column-split id
  const int tile = blockIdx.x;                  // 0 .. B*N/16 - 1
  const int b    = tile >> 8;                   // tile / (N/16)
  const int row0 = (tile & 255) << 4;

  const int h  = lane >> 4;                     // half-wave select
  const int ln = lane & 15;

  const unsigned short* Eb = Ebf + (size_t)b * Nn * Dn;

  // A fragments: contiguous K order chosen here (k = 16h + s): lanes 0-15 hold K {0..15},
  // lanes 16-31 hold K {16..31} for row M = lane&15 (matches B).
  V16 a[8];
  {
    const unsigned short* ar = Eb + (size_t)(row0 + ln) * Dn;
#pragma unroll
    for (int c = 0; c < 8; ++c) {
      const unsigned short* p = ar + c * 32 + h * 16;
      a[c].h[0] = *(const v8bf*)(p);
      a[c].h[1] = *(const v8bf*)(p + 8);
    }
  }

  // positive index for each of the 8 rows this lane's C slots cover
  int jr[8];
#pragma unroll
  for (int v = 0; v < 8; ++v) jr[v] = jpos[b * Nn + row0 + v + 8 * h];

  float m[8], s[8], pos[8];
#pragma unroll
  for (int v = 0; v < 8; ++v) {
    m[v] = -__builtin_inff(); s[v] = 0.f; pos[v] = -__builtin_inff();
  }

  auto load_tile = [&](int col0, v16bf* bt) {
    const unsigned short* br = Eb + (size_t)(col0 + ln) * Dn + h * 16;
#pragma unroll
    for (int c = 0; c < 8; ++c) bt[c] = *(const v16bf*)(br + c * 32);
  };
  auto chain = [&](const v16bf* bt) {
    v8f acc = {};
#pragma unroll
    for (int c = 0; c < 8; ++c) {
      acc = __builtin_amdgcn_wmma_f32_16x16x32_bf16(
          false, a[c].v, false, bt[c], (short)0, acc, false, false);
    }
    return acc;
  };
  auto online_update = [&](const v8f& acc, int col) {
#pragma unroll
    for (int v = 0; v < 8; ++v) {
      float x  = acc[v] * SCALE2;               // exp2-domain logits
      float mn = fmaxf(m[v], x);
      s[v] = s[v] * fast_exp2(m[v] - mn) + fast_exp2(x - mn);
      m[v] = mn;
      pos[v] = (col == jr[v]) ? x : pos[v];
    }
  };

  const int cbeg = wave * NCOLS;
  const int cend = cbeg + NCOLS;

  // prologue: first tile
  v16bf bt[8];
  load_tile(cbeg, bt);
  sched_fence();                 // keep all 16 b128 loads issued up front
  v8f accPrev = chain(bt);
  int colPrev = cbeg;

  for (int col0 = cbeg + 16; col0 < cend; col0 += 16) {
    load_tile(col0, bt);
    sched_fence();               // loads stay above; softmax may fill the chain
    v8f acc = chain(bt);
    online_update(accPrev, colPrev + ln);   // previous tile: independent work
    accPrev = acc;
    colPrev = col0;
  }
  online_update(accPrev, colPrev + ln);     // epilogue

  // combine (m,s,pos) across the 16 lanes of each half via ds_swizzle XOR
#define RED_STEP(PAT)                                                       \
  {                                                                         \
    _Pragma("unroll") for (int v = 0; v < 8; ++v) {                         \
      float mo = swz<PAT>(m[v]);                                            \
      float so = swz<PAT>(s[v]);                                            \
      float po = swz<PAT>(pos[v]);                                          \
      float mn = fmaxf(m[v], mo);                                           \
      s[v] = s[v] * fast_exp2(m[v] - mn) + so * fast_exp2(mo - mn);         \
      m[v] = mn;                                                            \
      pos[v] = fmaxf(pos[v], po);                                           \
    }                                                                       \
  }
  RED_STEP(0x041F)   // xor 1
  RED_STEP(0x081F)   // xor 2
  RED_STEP(0x101F)   // xor 4
  RED_STEP(0x201F)   // xor 8
#undef RED_STEP

  // per-wave partials -> LDS
  if (ln == 0) {
#pragma unroll
    for (int v = 0; v < 8; ++v) {
      const int r = v + 8 * h;
      sm[wave][r] = m[v];
      ss[wave][r] = s[v];
      sp[wave][r] = pos[v];
    }
  }
  __syncthreads();

  // one thread per row merges the CSPLIT partial softmax states
  if (threadIdx.x < 16) {
    const int r = threadIdx.x;
    float M = sm[0][r], S = ss[0][r], P = sp[0][r];
#pragma unroll
    for (int w = 1; w < CSPLIT; ++w) {
      float mo = sm[w][r], so = ss[w][r];
      float mn = fmaxf(M, mo);
      S = S * fast_exp2(M - mn) + so * fast_exp2(mo - mn);
      M = mn;
      P = fmaxf(P, sp[w][r]);
    }
    const int j = jpos[b * Nn + row0 + r];
    float loss = (j >= 0) ? LN2 * (M + fast_log2(S) - P) : 0.f;
    lossw[b * Nn + row0 + r] = loss;
  }
}

// -------------------------- 4) final reduction -----------------------------
__global__ __launch_bounds__(256)
void infonce_reduce_kernel(const float* __restrict__ lossw,
                           const int* __restrict__ jpos,
                           float* __restrict__ out) {
  __shared__ float ssum[256];
  __shared__ int   scnt[256];
  float total = 0.f;
  for (int b = 0; b < Bn; ++b) {
    float ls = 0.f; int lc = 0;
    for (int i = threadIdx.x; i < Nn; i += 256) {
      ls += lossw[b * Nn + i];
      lc += (jpos[b * Nn + i] >= 0) ? 1 : 0;
    }
    ssum[threadIdx.x] = ls; scnt[threadIdx.x] = lc;
    __syncthreads();
    for (int st = 128; st > 0; st >>= 1) {
      if (threadIdx.x < st) {
        ssum[threadIdx.x] += ssum[threadIdx.x + st];
        scnt[threadIdx.x] += scnt[threadIdx.x + st];
      }
      __syncthreads();
    }
    if (threadIdx.x == 0 && scnt[0] > 0) total += ssum[0] / (float)scnt[0];
    __syncthreads();
  }
  if (threadIdx.x == 0) out[0] = total / (float)Bn;
}

// ---------------------------------------------------------------------------
extern "C" void kernel_launch(void* const* d_in, const int* in_sizes, int n_in,
                              void* d_out, int out_size, void* d_ws, size_t ws_size,
                              hipStream_t stream) {
  (void)in_sizes; (void)n_in; (void)out_size; (void)ws_size;
  const float* emb    = (const float*)d_in[0];
  const int*   groups = (const int*)d_in[1];
  float*       out    = (float*)d_out;

  char* ws = (char*)d_ws;
  unsigned short* ebf = (unsigned short*)ws;                        // 8 MB bf16
  int*   jpos  = (int*)(ws + (size_t)Bn * Nn * Dn * 2);             // 64 KB
  float* lossw = (float*)(ws + (size_t)Bn * Nn * Dn * 2 + (size_t)Bn * Nn * 4);

  infonce_cvt_kernel<<<(Bn * Nn * Dn) / 1024, 256, 0, stream>>>(emb, ebf);
  infonce_pos_kernel<<<Bn * (Nn / 256), 256, 0, stream>>>(groups, jpos);
  infonce_gemm_kernel<<<Bn * Nn / 16, 128, 0, stream>>>(ebf, jpos, lossw);
  infonce_reduce_kernel<<<1, 256, 0, stream>>>(lossw, jpos, out);
}
